// MultiHeadedAttention_73899207295412
// MI455X (gfx1250) — matrix operand order repaired and checked by reading, not compiled, not run on hardware
//
#include <hip/hip_runtime.h>

typedef __attribute__((ext_vector_type(16))) _Float16 v16h;
typedef __attribute__((ext_vector_type(8)))  _Float16 v8h;
typedef __attribute__((ext_vector_type(8)))  float    v8f;

#define DMODEL 512
#define NHEAD  8
#define DK     64
#define BATCH  2
#define SEQ    4096

static __device__ inline v8f wmma16(v16h a, v16h b, v8f c) {
  return __builtin_amdgcn_wmma_f32_16x16x32_f16(false, a, false, b, (short)0, c,
                                                false, false);
}

// A-fragment: two 8-element contiguous f16 chunks -> v16h
static __device__ inline v16h ld8x2h(const _Float16* p0, const _Float16* p1) {
  v8h a = *(const v8h*)p0;
  v8h b = *(const v8h*)p1;
  return __builtin_shufflevector(a, b, 0, 1, 2, 3, 4, 5, 6, 7,
                                       8, 9, 10, 11, 12, 13, 14, 15);
}

// Async-copy one 4KB tile (32 rows x 64 f16) global -> LDS; 8 x b128 per wave,
// each lane moves 16 bytes per op. Tracked by ASYNCcnt.
static __device__ inline void async_copy_tile(const _Float16* gbase,
                                              unsigned ldsbase, int lane) {
#pragma unroll
  for (int i = 0; i < 8; ++i) {
    const char* g = (const char*)gbase + i * 512 + lane * 16;
    unsigned l = ldsbase + i * 512 + lane * 16;
    asm volatile("global_load_async_to_lds_b128 %0, %1, off"
                 :: "v"(l), "v"(g) : "memory");
  }
}

// ---------------------------------------------------------------------------
// Elementwise f32 -> f16 conversion, 8 elements/thread, exact-sized launches.
// ---------------------------------------------------------------------------
__global__ void __launch_bounds__(256)
cvt_kernel(const float* __restrict__ src, _Float16* __restrict__ dst) {
  const size_t i = ((size_t)blockIdx.x * 256 + threadIdx.x) * 8;
  const float4 u0 = *(const float4*)(src + i);
  const float4 u1 = *(const float4*)(src + i + 4);
  v8h h;
  h[0] = (_Float16)u0.x; h[1] = (_Float16)u0.y;
  h[2] = (_Float16)u0.z; h[3] = (_Float16)u0.w;
  h[4] = (_Float16)u1.x; h[5] = (_Float16)u1.y;
  h[6] = (_Float16)u1.z; h[7] = (_Float16)u1.w;
  *(v8h*)(dst + i) = h;
}

// ---------------------------------------------------------------------------
// Projection: Y[m,n] = sum_k Xc[m,k]*Wc[n,k] + bias[n]  (all-f16 fragments)
// Each wave: 32 rows x 64 cols; k-loop fully unrolled (16 steps, 8 WMMAs/step).
// dst: vtrans==0 -> [B,H,L,dk], vtrans==1 -> [B,H,dk,L]
// grid (64, 8), block 128 (4 waves)
// ---------------------------------------------------------------------------
__global__ void __launch_bounds__(128)
proj16_kernel(const _Float16* __restrict__ Xc, const _Float16* __restrict__ Wc,
              const float* __restrict__ bias, _Float16* __restrict__ dst,
              int vtrans) {
  const int lane  = threadIdx.x & 31;
  const int wave  = threadIdx.x >> 5;
  const int mtile = (blockIdx.x * 4 + wave) * 32;
  const int n0    = blockIdx.y * 64;
  const int mrow  = lane & 15;
  const int c     = lane >> 4;

  const _Float16* arow0 = Xc + (size_t)(mtile + mrow) * DMODEL;
  const _Float16* arow1 = arow0 + 16 * DMODEL;
  const _Float16* wrow[4];
#pragma unroll
  for (int t = 0; t < 4; ++t)
    wrow[t] = Wc + (size_t)(n0 + 16 * t + mrow) * DMODEL + 16 * c;

  v8f acc[2][4] = {};
#pragma unroll
  for (int k0 = 0; k0 < DMODEL; k0 += 32) {
    v16h a0 = ld8x2h(arow0 + k0 + 16 * c, arow0 + k0 + 16 * c + 8);
    v16h a1 = ld8x2h(arow1 + k0 + 16 * c, arow1 + k0 + 16 * c + 8);
#pragma unroll
    for (int t = 0; t < 4; ++t) {
      const v16h b = *(const v16h*)(wrow[t] + k0);
      acc[0][t] = wmma16(a0, b, acc[0][t]);
      acc[1][t] = wmma16(a1, b, acc[1][t]);
    }
  }

#pragma unroll
  for (int t = 0; t < 4; ++t) {
    const int n = n0 + 16 * t + mrow;
    const float bv = bias[n];
    const int h = n >> 6, d = n & 63;
#pragma unroll
    for (int s = 0; s < 2; ++s) {
#pragma unroll
      for (int r = 0; r < 8; ++r) {
        const int m    = mtile + s * 16 + r + 8 * c;
        const int bidx = m >> 12;
        const int pos  = m & 4095;
        const float val = acc[s][t][r] + bv;
        if (vtrans)
          dst[((size_t)(bidx * NHEAD + h) * DK + d) * SEQ + pos] = (_Float16)val;
        else
          dst[((size_t)(bidx * NHEAD + h) * SEQ + pos) * DK + d] = (_Float16)val;
      }
    }
  }
}

// ---------------------------------------------------------------------------
// Flash attention: per wave, one 16-query tile; online softmax over 32-key
// blocks. K-tiles are staged global->LDS with async b128 copies (ASYNCcnt),
// double-buffered, one full tile of prefetch distance. V fragments direct.
// Qh,Kh: [B,H,L,dk] f16; Vt: [B,H,dk,L] f16; Ho: [B*L, 512] f16.
// grid (64, 16), block 128 (4 waves).
// ---------------------------------------------------------------------------
__global__ void __launch_bounds__(128)
attn_kernel(const _Float16* __restrict__ Qh, const _Float16* __restrict__ Kh,
            const _Float16* __restrict__ Vt, _Float16* __restrict__ Ho) {
  __shared__ __attribute__((aligned(32))) _Float16 kbuf[4][2][32 * 64]; // 32KB
  __shared__ __attribute__((aligned(32))) _Float16 pbuf[4][16 * 32];    //  4KB
  const int lane = threadIdx.x & 31;
  const int wave = threadIdx.x >> 5;
  const int bh   = blockIdx.y;           // b*8 + h
  const int qt   = blockIdx.x * 4 + wave;
  const int mrow = lane & 15;
  const int c    = lane >> 4;

  const _Float16* Qb = Qh + (size_t)bh * SEQ * DK;
  const _Float16* Kb = Kh + (size_t)bh * SEQ * DK;
  const _Float16* Vb = Vt + (size_t)bh * DK * SEQ;
  _Float16* P = pbuf[wave];
  const unsigned klds[2] = {
      (unsigned)(uintptr_t)&kbuf[wave][0][0],
      (unsigned)(uintptr_t)&kbuf[wave][1][0]};

  const _Float16* qrow = Qb + (size_t)(qt * 16 + mrow) * DK;
  const v16h qa0 = ld8x2h(qrow + 16 * c, qrow + 16 * c + 8);       // dk 0..31
  const v16h qa1 = ld8x2h(qrow + 32 + 16 * c, qrow + 40 + 16 * c);  // dk 32..63

  const _Float16* vr0 = Vb + (size_t)mrow * SEQ + 16 * c;         // Vt row (d)

  float rmax[8], rsum[8];
#pragma unroll
  for (int r = 0; r < 8; ++r) { rmax[r] = -1e30f; rsum[r] = 0.f; }
  v8f acc[4] = {};

  // prime the pipeline: tile 0 -> buffer 0
  async_copy_tile(Kb, klds[0], lane);

  for (int kb = 0; kb < SEQ; kb += 32) {
    const int cur = (kb >> 5) & 1;
    // issue next K-tile into the alternate buffer, then wait so that the
    // 8 older ops (current tile) are done while 8 newer stay in flight.
    if (kb + 32 < SEQ) {
      async_copy_tile(Kb + (size_t)(kb + 32) * DK, klds[cur ^ 1], lane);
      asm volatile("s_wait_asynccnt 0x8" ::: "memory");
      __builtin_prefetch(vr0 + kb + 32, 0, 1);
    } else {
      asm volatile("s_wait_asynccnt 0x0" ::: "memory");
    }
    // scores S(16 x 32) = q . k^T : 4 WMMAs, K fragments from LDS
    const _Float16* kl = &kbuf[wave][cur][0];
    v8f s0 = {}, s1 = {};
    {
      v16h b00 = *(const v16h*)(kl + mrow * DK + 16 * c);
      v16h b01 = *(const v16h*)(kl + mrow * DK + 32 + 16 * c);
      v16h b10 = *(const v16h*)(kl + (16 + mrow) * DK + 16 * c);
      v16h b11 = *(const v16h*)(kl + (16 + mrow) * DK + 32 + 16 * c);
      s0 = wmma16(qa0, b00, s0);
      s0 = wmma16(qa1, b01, s0);
      s1 = wmma16(qa0, b10, s1);
      s1 = wmma16(qa1, b11, s1);
    }
    // online softmax on C-layout; row m = r + 8*c lives in one VGPR across a
    // 16-lane half-wave -> half-wave shuffle reductions.
#pragma unroll
    for (int r = 0; r < 8; ++r) {
      float a0 = s0[r] * 0.125f;   // 1/sqrt(dk)
      float a1 = s1[r] * 0.125f;
      float mx = fmaxf(a0, a1);
#pragma unroll
      for (int off = 1; off < 16; off <<= 1)
        mx = fmaxf(mx, __shfl_xor(mx, off, 32));
      const float mnew = fmaxf(rmax[r], mx);
      const float sc = __expf(rmax[r] - mnew);
      const float e0 = __expf(a0 - mnew);
      const float e1 = __expf(a1 - mnew);
      float ps = e0 + e1;
#pragma unroll
      for (int off = 1; off < 16; off <<= 1)
        ps += __shfl_xor(ps, off, 32);
      rsum[r] = rsum[r] * sc + ps;
      rmax[r] = mnew;
#pragma unroll
      for (int t = 0; t < 4; ++t) acc[t][r] *= sc;
      const int m = r + 8 * c;
      P[m * 32 + mrow]      = (_Float16)e0;
      P[m * 32 + 16 + mrow] = (_Float16)e1;
    }
    // P is wave-private: wait for this wave's ds-stores, then reload in
    // A-fragment order. (LDS ops of one wave are in-order; no WG barrier.)
    asm volatile("s_wait_dscnt 0x0" ::: "memory");
    const _Float16* prow = P + mrow * 32;
    v16h pa = ld8x2h(prow + 16 * c, prow + 16 * c + 8);
    // O(16 x 64) += P(16x32) . V(32x64) : 4 WMMAs
#pragma unroll
    for (int t = 0; t < 4; ++t) {
      v16h bv = *(const v16h*)(vr0 + (size_t)t * 16 * SEQ + kb);
      acc[t] = wmma16(pa, bv, acc[t]);
    }
    asm volatile("s_wait_dscnt 0x0" ::: "memory");
  }
  // finalize: divide by row sums, write heads concatenated -> [B*L, 512]
  const int b = bh >> 3, h = bh & 7;
#pragma unroll
  for (int t = 0; t < 4; ++t) {
#pragma unroll
    for (int r = 0; r < 8; ++r) {
      const int m = qt * 16 + r + 8 * c;
      const float o = acc[t][r] / rsum[r];
      Ho[(size_t)(b * SEQ + m) * DMODEL + h * DK + t * 16 + mrow] = (_Float16)o;
    }
  }
}

// ---------------------------------------------------------------------------
// Output projection: out = Ho . Wo^T + bo  (f16 fragments, f32 out)
// Each wave: 32 rows x 64 cols; fully unrolled k-loop. grid (64, 8).
// ---------------------------------------------------------------------------
__global__ void __launch_bounds__(128)
oproj_kernel(const _Float16* __restrict__ A, const _Float16* __restrict__ Wc,
             const float* __restrict__ bias, float* __restrict__ out) {
  const int lane  = threadIdx.x & 31;
  const int wave  = threadIdx.x >> 5;
  const int mtile = (blockIdx.x * 4 + wave) * 32;
  const int n0    = blockIdx.y * 64;
  const int mrow  = lane & 15;
  const int c     = lane >> 4;

  const _Float16* arow0 = A + (size_t)(mtile + mrow) * DMODEL;
  const _Float16* arow1 = arow0 + 16 * DMODEL;
  const _Float16* wrow[4];
#pragma unroll
  for (int t = 0; t < 4; ++t)
    wrow[t] = Wc + (size_t)(n0 + 16 * t + mrow) * DMODEL + 16 * c;

  v8f acc[2][4] = {};
#pragma unroll
  for (int k0 = 0; k0 < DMODEL; k0 += 32) {
    v16h a0 = ld8x2h(arow0 + k0 + 16 * c, arow0 + k0 + 16 * c + 8);
    v16h a1 = ld8x2h(arow1 + k0 + 16 * c, arow1 + k0 + 16 * c + 8);
#pragma unroll
    for (int t = 0; t < 4; ++t) {
      const v16h b = *(const v16h*)(wrow[t] + k0);
      acc[0][t] = wmma16(a0, b, acc[0][t]);
      acc[1][t] = wmma16(a1, b, acc[1][t]);
    }
  }

#pragma unroll
  for (int t = 0; t < 4; ++t) {
    const int n = n0 + 16 * t + mrow;
    const float bv = bias[n];
#pragma unroll
    for (int s = 0; s < 2; ++s) {
#pragma unroll
      for (int r = 0; r < 8; ++r) {
        const int m = mtile + s * 16 + r + 8 * c;
        out[(size_t)m * DMODEL + n] = acc[s][t][r] + bv;
      }
    }
  }
}

extern "C" void kernel_launch(void* const* d_in, const int* in_sizes, int n_in,
                              void* d_out, int out_size, void* d_ws, size_t ws_size,
                              hipStream_t stream) {
  const float* Q  = (const float*)d_in[0];
  const float* K  = (const float*)d_in[1];
  const float* V  = (const float*)d_in[2];
  const float* Wq = (const float*)d_in[3];
  const float* bq = (const float*)d_in[4];
  const float* Wk = (const float*)d_in[5];
  const float* bk = (const float*)d_in[6];
  const float* Wv = (const float*)d_in[7];
  const float* bv = (const float*)d_in[8];
  const float* Wo = (const float*)d_in[9];
  const float* bo = (const float*)d_in[10];
  float* out = (float*)d_out;

  _Float16* ws = (_Float16*)d_ws;
  const size_t perBuf = (size_t)BATCH * NHEAD * SEQ * DK;  // 4M f16
  const size_t wBuf   = (size_t)DMODEL * DMODEL;           // 256K f16
  _Float16* Qh  = ws;
  _Float16* Kh  = ws + 1 * perBuf;
  _Float16* Vt  = ws + 2 * perBuf;
  _Float16* Ho  = ws + 3 * perBuf;
  _Float16* Xq  = ws + 4 * perBuf;
  _Float16* Xk  = ws + 5 * perBuf;
  _Float16* Xv  = ws + 6 * perBuf;
  _Float16* Wqc = ws + 7 * perBuf;
  _Float16* Wkc = Wqc + wBuf;
  _Float16* Wvc = Wkc + wBuf;
  _Float16* Woc = Wvc + wBuf;

  // f32 -> f16 staging (HBM-bound)
  const int gX = (int)(perBuf / 8 / 256);  // 2048 blocks
  const int gW = (int)(wBuf / 8 / 256);    // 128 blocks
  cvt_kernel<<<gX, 256, 0, stream>>>(Q, Xq);
  cvt_kernel<<<gX, 256, 0, stream>>>(K, Xk);
  cvt_kernel<<<gX, 256, 0, stream>>>(V, Xv);
  cvt_kernel<<<gW, 256, 0, stream>>>(Wq, Wqc);
  cvt_kernel<<<gW, 256, 0, stream>>>(Wk, Wkc);
  cvt_kernel<<<gW, 256, 0, stream>>>(Wv, Wvc);
  cvt_kernel<<<gW, 256, 0, stream>>>(Wo, Woc);

  dim3 blk(128);
  dim3 gproj(64, 8);
  proj16_kernel<<<gproj, blk, 0, stream>>>(Xq, Wqc, bq, Qh, 0);
  proj16_kernel<<<gproj, blk, 0, stream>>>(Xk, Wkc, bk, Kh, 0);
  proj16_kernel<<<gproj, blk, 0, stream>>>(Xv, Wvc, bv, Vt, 1);
  dim3 gattn(64, 16);
  attn_kernel<<<gattn, blk, 0, stream>>>(Qh, Kh, Vt, Ho);
  oproj_kernel<<<gproj, blk, 0, stream>>>(Ho, Woc, bo, out);
}
